// MLPQ_91001767068432
// MI455X (gfx1250) — 8-bit matrix operand order repaired and checked by reading, compiled, not run on hardware
//
#include <hip/hip_runtime.h>

typedef __attribute__((ext_vector_type(8))) int v8i;

#define LOG2E 1.4426950408889634f
#define INV127SQ (1.0f / 16129.0f)

__device__ __forceinline__ float q8f(float x, float s) {
  float r = rintf(x * s);
  return fminf(fmaxf(r, -127.0f), 127.0f);
}
__device__ __forceinline__ int packq4(float a, float b, float c, float d, float s) {
  unsigned ia = (unsigned)(int)q8f(a, s), ib = (unsigned)(int)q8f(b, s);
  unsigned ic = (unsigned)(int)q8f(c, s), id = (unsigned)(int)q8f(d, s);
  unsigned lo = __builtin_amdgcn_perm(ib, ia, 0x0c0c0400u);
  unsigned hi = __builtin_amdgcn_perm(id, ic, 0x0c0c0400u);
  return (int)__builtin_amdgcn_perm(hi, lo, 0x05040100u);
}
__device__ __forceinline__ float fast_sigmoid(float x) {
  return __builtin_amdgcn_rcpf(1.0f + __builtin_amdgcn_exp2f(-LOG2E * x));
}
__device__ __forceinline__ float fast_tanh(float x) {
  return 2.0f * __builtin_amdgcn_rcpf(1.0f + __builtin_amdgcn_exp2f(-2.0f * LOG2E * x)) - 1.0f;
}
__device__ __forceinline__ float qscale(float amax) {
  return 127.0f * __builtin_amdgcn_rcpf(amax);
}

__global__ void pack_weights(const float* __restrict__ w1,
                             const float* __restrict__ w_ih,
                             const float* __restrict__ w_hh,
                             const float* __restrict__ w_out,
                             signed char* __restrict__ ws) {
  int idx = blockIdx.x * blockDim.x + threadIdx.x;
  if (idx < 2048) {
    int kh = idx >> 10, rem = idx & 1023;
    int lane = rem >> 5, b = rem & 31;
    int v = b >> 2, j = b & 3;
    int n = lane & 15;
    int K = kh * 64 + (v >= 4 ? 32 : 0) + ((lane >> 4) ? 16 : 0) + (v & 3) * 4 + j;
    ws[idx] = (signed char)(int)w1[n * 128 + K];
  } else if (idx < 4096) {
    int id = idx - 2048;
    int t = id >> 9, rem = id & 511;
    int lane = rem >> 4, j = rem & 15;
    int ch = 16 * t + (lane & 15);
    float val = (j < 8) ? w_ih[ch * 16 + (lane >> 4) * 8 + j] : w_hh[ch * 16 + (lane >> 4) * 8 + j - 8];
    ws[idx] = (signed char)(int)val;
  } else if (idx < 4608) {
    int id = idx - 4096;
    int lane = id >> 4, j = id & 15;
    int n = lane & 15;
    float val = (n < 4 && j < 8) ? w_out[n * 16 + (lane >> 4) * 8 + j] : 0.0f;
    ws[idx] = (signed char)(int)val;
  }
}

__global__ void __launch_bounds__(256)
mlstm_fused(const float* __restrict__ x, const float* __restrict__ h,
            const float* __restrict__ c,
            const signed char* __restrict__ ws,
            const float* __restrict__ amax_in0, const float* __restrict__ amax_w0,
            const float* __restrict__ amax_lstm_in, const float* __restrict__ amax_w_lstm,
            const float* __restrict__ amax_out_in, const float* __restrict__ amax_w_out,
            float* __restrict__ xo_out, float* __restrict__ hy_out,
            float* __restrict__ cy_out) {
  const int lane = threadIdx.x & 31;
  const int wave = threadIdx.x >> 5;
  const int m    = lane & 15;
  const int half = lane >> 4;
  const int row0 = (blockIdx.x * 8 + wave) * 16;

  __shared__ signed char lds[8][256];
  signed char* tbuf = lds[wave];

  const float a_in0  = amax_in0[0];
  const float a_lstm = amax_lstm_in[0];
  const float a_out  = amax_out_in[0];
  const float qs_x = qscale(a_in0);
  const float qs_h = qscale(a_lstm);
  const float qs_o = qscale(a_out);

  v8i A0 = {}, A1 = {};
  const float* xr = x + (size_t)(row0 + m) * 128;
#pragma unroll
  for (int v = 0; v < 8; ++v) {
    const int k = ((v >> 2) * 32) + half * 16 + (v & 3) * 4;
    float4 f0 = *(const float4*)(xr + k);
    float4 f1 = *(const float4*)(xr + 64 + k);
    ((int*)&A0)[v] = packq4(f0.x, f0.y, f0.z, f0.w, qs_x);
    ((int*)&A1)[v] = packq4(f1.x, f1.y, f1.z, f1.w, qs_x);
  }

  v8i Bw0 = {}, Bw1 = {};
  {
    const int4* p = (const int4*)ws;
    int4 lo0 = p[lane * 2 + 0], hi0 = p[lane * 2 + 1];
    int4 lo1 = p[64 + lane * 2 + 0], hi1 = p[64 + lane * 2 + 1];
    Bw0[0] = lo0.x; Bw0[1] = lo0.y; Bw0[2] = lo0.z; Bw0[3] = lo0.w;
    Bw0[4] = hi0.x; Bw0[5] = hi0.y; Bw0[6] = hi0.z; Bw0[7] = hi0.w;
    Bw1[0] = lo1.x; Bw1[1] = lo1.y; Bw1[2] = lo1.z; Bw1[3] = lo1.w;
    Bw1[4] = hi1.x; Bw1[5] = hi1.y; Bw1[6] = hi1.z; Bw1[7] = hi1.w;
  }

  v8i acc = {};
  acc = __builtin_amdgcn_wmma_i32_16x16x64_iu8(true, A0, true, Bw0, acc, false, false);
  acc = __builtin_amdgcn_wmma_i32_16x16x64_iu8(true, A1, true, Bw1, acc, false, false);

  {
    const float dq0 = (a_in0 * amax_w0[m]) * INV127SQ;
#pragma unroll
    for (int r = 0; r < 8; ++r) {
      float af = fmaxf((float)acc[r], 0.0f) * dq0;
      tbuf[(half * 8 + r) * 16 + m] = (signed char)(int)q8f(af, qs_h);
    }
  }
  asm volatile("s_wait_dscnt 0" ::: "memory");

  v8i A2 = {};
  {
    int2 av = *(const int2*)&tbuf[m * 16 + half * 8];
    A2[0] = av.x; A2[1] = av.y;
    const float* hr = h + (size_t)(row0 + m) * 16 + half * 8;
    float4 h0 = *(const float4*)(hr);
    float4 h1 = *(const float4*)(hr + 4);
    A2[2] = packq4(h0.x, h0.y, h0.z, h0.w, qs_h);
    A2[3] = packq4(h1.x, h1.y, h1.z, h1.w, qs_h);
  }

  v8i G[4];
#pragma unroll
  for (int t = 0; t < 4; ++t) {
    int4 bg = ((const int4*)(ws + 2048))[t * 32 + lane];
    v8i Bt = {};
    Bt[0] = bg.x; Bt[1] = bg.y; Bt[2] = bg.z; Bt[3] = bg.w;
    v8i z = {};
    G[t] = __builtin_amdgcn_wmma_i32_16x16x64_iu8(true, A2, true, Bt, z, false, false);
  }

  float amax_ch[4];
#pragma unroll
  for (int t = 0; t < 4; ++t) amax_ch[t] = a_lstm * amax_w_lstm[16 * t + m];

  float gq[4][8];
#pragma unroll
  for (int t = 0; t < 4; ++t) {
    const float dq = amax_ch[t] * INV127SQ;
    const float qs = qscale(amax_ch[t]);
#pragma unroll
    for (int r = 0; r < 8; ++r) {
      float gf = (float)G[t][r] * dq;
      gf = (t == 2) ? fast_tanh(gf) : fast_sigmoid(gf);
      gq[t][r] = rintf(gf * qs);
    }
  }

  const float dq_fc = (amax_ch[1] * 3.4764f) * INV127SQ;
  const float dq_ig = (amax_ch[0] * amax_ch[2]) * INV127SQ;
  const float dq_o  = amax_ch[3] * (1.0f / 127.0f);
  const size_t rbase = (size_t)(row0 + half * 8);
  const float* cp = c      + rbase * 16 + m;
  float*      cyp = cy_out + rbase * 16 + m;
  float*      hyp = hy_out + rbase * 16 + m;
#pragma unroll
  for (int r = 0; r < 8; ++r) {
    float cqv = q8f(cp[r * 16], qs_h);
    float cy  = (gq[1][r] * cqv) * dq_fc + (gq[0][r] * gq[2][r]) * dq_ig;
    float od  = gq[3][r] * dq_o;
    float hy  = od * fast_tanh(cy);
    cyp[r * 16] = cy;
    hyp[r * 16] = hy;
    tbuf[(half * 8 + r) * 16 + m] = (signed char)(int)q8f(hy, qs_o);
  }
  asm volatile("s_wait_dscnt 0" ::: "memory");

  v8i A3 = {};
  {
    int2 hv = *(const int2*)&tbuf[m * 16 + half * 8];
    A3[0] = hv.x; A3[1] = hv.y;
  }
  v8i B3 = {};
  {
    int4 bo = ((const int4*)(ws + 4096))[lane];
    B3[0] = bo.x; B3[1] = bo.y; B3[2] = bo.z; B3[3] = bo.w;
  }
  v8i zz = {};
  v8i O = __builtin_amdgcn_wmma_i32_16x16x64_iu8(true, A3, true, B3, zz, false, false);
  if (m < 4) {
    const float dqo = (a_out * amax_w_out[m]) * INV127SQ;
    float* xop = xo_out + rbase * 4 + m;
#pragma unroll
    for (int r = 0; r < 8; ++r) {
      xop[r * 4] = (float)O[r] * dqo;
    }
  }
}

extern "C" void kernel_launch(void* const* d_in, const int* in_sizes, int n_in,
                              void* d_out, int out_size, void* d_ws, size_t ws_size,
                              hipStream_t stream) {
  const float* x            = (const float*)d_in[0];
  const float* h            = (const float*)d_in[1];
  const float* c            = (const float*)d_in[2];
  const float* w1           = (const float*)d_in[3];
  const float* w_ih         = (const float*)d_in[4];
  const float* w_hh         = (const float*)d_in[5];
  const float* w_out        = (const float*)d_in[6];
  const float* amax_in0     = (const float*)d_in[7];
  const float* amax_w0      = (const float*)d_in[8];
  const float* amax_lstm_in = (const float*)d_in[9];
  const float* amax_w_lstm  = (const float*)d_in[10];
  const float* amax_out_in  = (const float*)d_in[11];
  const float* amax_w_out   = (const float*)d_in[12];

  const int B = in_sizes[0] / 128;
  signed char* ws = (signed char*)d_ws;

  float* xo_out = (float*)d_out;
  float* hy_out = xo_out + (size_t)B * 4;
  float* cy_out = hy_out + (size_t)B * 16;

  pack_weights<<<18, 256, 0, stream>>>(w1, w_ih, w_hh, w_out, ws);

  const int blocks = B / 128;
  mlstm_fused<<<blocks, 256, 0, stream>>>(x, h, c, ws,
                                          amax_in0, amax_w0, amax_lstm_in,
                                          amax_w_lstm, amax_out_in, amax_w_out,
                                          xo_out, hy_out, cy_out);
}
